// WaveletAttention_82214263980295
// MI455X (gfx1250) — matrix operand order repaired and checked by reading, not compiled, not run on hardware
//
#include <hip/hip_runtime.h>
#include <hip/hip_bf16.h>
#include <stdint.h>

// ---------------- problem constants (from reference) ----------------
#define S_   3
#define D_   128
#define H_   8
#define C_   64
#define B_   2
#define T_   2048
#define DH_  16
#define BT_  (B_*T_)

// db4 analysis filters (hardcoded from reference REC_LO)
__constant__ float DEC_LO_C[8] = {
  -0.010597401784997278f,  0.032883011666982945f,  0.030841381835986965f,
  -0.18703481171888114f,  -0.02798376941698385f,   0.6308807679295904f,
   0.7148465705525415f,    0.23037781330885523f };
__constant__ float DEC_HI_C[8] = {
  -0.23037781330885523f,   0.7148465705525415f,   -0.6308807679295904f,
  -0.02798376941698385f,   0.18703481171888114f,   0.030841381835986965f,
  -0.032883011666982945f, -0.010597401784997278f };

typedef _Float16 v16h __attribute__((ext_vector_type(16)));
typedef float    v8f  __attribute__((ext_vector_type(8)));

__device__ __forceinline__ v8f wmma_f16(v16h a, v16h b, v8f c) {
  // D = A(16x32 f16) * B(32x16 f16) + C(16x16 f32)
  return __builtin_amdgcn_wmma_f32_16x16x32_f16(false, a, false, b, (short)0, c,
                                                false, false);
}

// A-matrix (16x32 f16) loader from row-major src [16 rows x ld], cols kc..kc+31.
// Layout: lane<16 -> M=lane, halves 0..15 = K kc+0..15
//         lane>=16 -> M=lane-16, halves 0..15 = K kc+16..31
__device__ __forceinline__ v16h loadA16x32(const _Float16* __restrict__ src, int ld, int kc) {
  int l = threadIdx.x & 31;
  const _Float16* p = src + (size_t)(l & 15) * ld + kc + ((l & 16) ? 16 : 0);
  v16h a;
#pragma unroll
  for (int i = 0; i < 16; i++) a[i] = p[i];
  return a;
}

// B-matrix (32x16 f16) loader from "B^T" row-major srcT [16 n-rows x ld], cols kc..kc+31.
// Same contiguous k order as loadA16x32: lanes 0..15 hold K=kc+0..15 (halves 0..15), lanes 16..31 hold K=kc+16..31.
__device__ __forceinline__ v16h loadB32x16(const _Float16* __restrict__ srcT, int ld, int kc) {
  int l = threadIdx.x & 31;
  const _Float16* p = srcT + (size_t)(l & 15) * ld + kc + ((l & 16) ? 16 : 0);
  v16h b;
#pragma unroll
  for (int i = 0; i < 16; i++) b[i] = p[i];
  return b;
}

// K A-tile: M=16 keys, K=dh padded to 32 (lanes 16..31 zero)
__device__ __forceinline__ v16h loadK_A(const _Float16* __restrict__ kh, int keybase, int lane) {
  const _Float16* p = kh + (size_t)(keybase + (lane & 15)) * DH_;
  v16h a;
#pragma unroll
  for (int i = 0; i < 16; i++) a[i] = (lane & 16) ? (_Float16)0.f : p[i];
  return a;
}

// V^T A-tile: M=dh(16), K=32 keys — contiguous thanks to transposed V
__device__ __forceinline__ v16h loadV_A(const _Float16* __restrict__ vh, int c0, int lane) {
  const _Float16* p = vh + (size_t)(lane & 15) * T_ + c0 + ((lane & 16) ? 16 : 0);
  v16h a;
#pragma unroll
  for (int i = 0; i < 16; i++) a[i] = p[i];
  return a;
}

// Online-softmax update for one 16-query tile holding 32 fresh scores
// (s_lo = keys 0..15 of chunk, s_hi = keys 16..31, C-layout of S^T: lane = query,
// pair lane^16 holds the other 8 keys of each sub-tile). Returns P^T B-tile.
__device__ __forceinline__ v16h softmax_update(v8f s_lo, v8f s_hi,
                                               float& m_run, float& l_run,
                                               v8f& ot, int lane) {
  float mx = s_lo[0];
#pragma unroll
  for (int r = 1; r < 8; r++) mx = fmaxf(mx, s_lo[r]);
#pragma unroll
  for (int r = 0; r < 8; r++) mx = fmaxf(mx, s_hi[r]);
  mx = fmaxf(mx, __shfl_xor(mx, 16, 32));
  float m_new = fmaxf(m_run, mx);
  float rescale = __expf(m_run - m_new);

  v8f p_lo, p_hi;
  float rsum = 0.f;
#pragma unroll
  for (int r = 0; r < 8; r++) {
    p_lo[r] = __expf(s_lo[r] - m_new); rsum += p_lo[r];
    p_hi[r] = __expf(s_hi[r] - m_new); rsum += p_hi[r];
  }
  rsum += __shfl_xor(rsum, 16, 32);
  l_run = l_run * rescale + rsum;
  m_run = m_new;
#pragma unroll
  for (int r = 0; r < 8; r++) ot[r] *= rescale;

  // C-layout -> B-layout (K=32 keys, N=16 queries) with xor-16 shuffles only
  union { _Float16 h[8]; int i[4]; } u0, u1, w0, w1;
#pragma unroll
  for (int r = 0; r < 8; r++) { u0.h[r] = (_Float16)p_lo[r]; u1.h[r] = (_Float16)p_hi[r]; }
#pragma unroll
  for (int j = 0; j < 4; j++) {
    w0.i[j] = __shfl_xor(u0.i[j], 16, 32);
    w1.i[j] = __shfl_xor(u1.i[j], 16, 32);
  }
  v16h pb;
  if (lane < 16) {
#pragma unroll
    for (int r = 0; r < 8; r++) { pb[r] = u0.h[r]; pb[r + 8] = w0.h[r]; }
  } else {
#pragma unroll
    for (int r = 0; r < 8; r++) { pb[r] = w1.h[r]; pb[r + 8] = u1.h[r]; }
  }
  return pb;
}

// ---------------- 1) SWT level (a-trous, periodic) ----------------
__global__ void wa_swt_level(const float* __restrict__ a_in,
                             float* __restrict__ a_out,
                             _Float16* __restrict__ det_out, int step) {
  int idx = blockIdx.x * blockDim.x + threadIdx.x;       // over B*T*C
  if (idx >= B_ * T_ * C_) return;
  int c = idx % C_;
  int t = (idx / C_) % T_;
  int b = idx / (C_ * T_);
  float dlo = 0.f, dhi = 0.f;
#pragma unroll
  for (int l = 0; l < 8; l++) {
    int tt = (t - l * step) & (T_ - 1);                  // T is pow2 -> mod
    float v = a_in[((size_t)b * T_ + tt) * C_ + c];
    dlo += v * DEC_LO_C[l];
    dhi += v * DEC_HI_C[l];
  }
  a_out[idx] = dlo;
  det_out[idx] = (_Float16)dhi;
}

// ---------------- 2) fold in_w into per-scale Q/K/V weights ----------------
__global__ void wa_fold_weights(const float* __restrict__ Wq, const float* __restrict__ bq,
                                const float* __restrict__ Wk, const float* __restrict__ bk,
                                const float* __restrict__ Wv, const float* __restrict__ bv,
                                const float* __restrict__ in_w, const float* __restrict__ in_b,
                                _Float16* __restrict__ Aout, float* __restrict__ bout) {
  int bid = blockIdx.x;              // 3*S*D blocks
  int n = bid % D_;
  int s = (bid / D_) % S_;
  int p = bid / (D_ * S_);
  const float* Wx = (p == 0) ? Wq : ((p == 1) ? Wk : Wv);
  const float* bx = (p == 0) ? bq : ((p == 1) ? bk : bv);
  const float* iw = in_w + ((size_t)p * D_ + n) * D_;
  int c = threadIdx.x;               // 64 threads
  float acc = 0.f;
  for (int d = 0; d < D_; d++) acc += iw[d] * Wx[((size_t)s * D_ + d) * C_ + c];
  Aout[(((size_t)p * S_ + s) * D_ + n) * C_ + c] = (_Float16)acc;
  if (c == 0) {
    float ab = 0.f;
    for (int d = 0; d < D_; d++) ab += iw[d] * bx[(size_t)s * D_ + d];
    bout[((size_t)p * S_ + s) * D_ + n] = ab + in_b[(size_t)p * D_ + n];
  }
}

__global__ void wa_cvt_outw(const float* __restrict__ out_w, _Float16* __restrict__ owh) {
  int i = blockIdx.x * blockDim.x + threadIdx.x;
  if (i < D_ * D_) owh[i] = (_Float16)out_w[i];
}

// ---------------- 3) QKV GEMM (WMMA): [tokens x 64] x [64 x 128] ----------------
__global__ void wa_qkv_gemm(const _Float16* __restrict__ Wdet,   // [S][B*T][C]
                            const _Float16* __restrict__ Amat,   // [3][S][D][C]
                            const float* __restrict__ bias,      // [3][S][D]
                            _Float16* __restrict__ q_out,        // [S][B][H][T][16]
                            _Float16* __restrict__ k_out,        // [S][B][H][T][16]
                            _Float16* __restrict__ vT_out) {     // [S][B][H][16][T]
  int wid = blockIdx.x * 8 + (threadIdx.x >> 5);   // blockDim 256
  int ntile = wid & 7;
  int ttile = (wid >> 3) & (BT_ / 16 - 1);
  int s = wid >> 11;
  if (s >= S_) return;
  int lane = threadIdx.x & 31;

  const _Float16* Wsrc = Wdet + ((size_t)s * BT_ + (size_t)ttile * 16) * C_;
  v16h a0 = loadA16x32(Wsrc, C_, 0);
  v16h a1 = loadA16x32(Wsrc, C_, 32);

  int dh = lane & 15;
  int n = ntile * 16 + dh;
  int rowoff = (lane & 16) ? 8 : 0;
  int b_idx = (ttile * 16) >> 11;        // token/T
  int tbase = (ttile * 16) & (T_ - 1);
  size_t head_base = ((size_t)s * B_ + b_idx) * H_ + ntile;

#pragma unroll
  for (int p = 0; p < 3; p++) {
    const _Float16* Bsrc = Amat + (((size_t)p * S_ + s) * D_ + (size_t)ntile * 16) * C_;
    v16h b0 = loadB32x16(Bsrc, C_, 0);
    v16h b1 = loadB32x16(Bsrc, C_, 32);
    v8f acc;
#pragma unroll
    for (int r = 0; r < 8; r++) acc[r] = 0.f;
    acc = wmma_f16(a0, b0, acc);
    acc = wmma_f16(a1, b1, acc);
    float bb = bias[((size_t)p * S_ + s) * D_ + n];
    if (p < 2) {
      _Float16* dst = ((p == 0) ? q_out : k_out) + head_base * T_ * DH_;
#pragma unroll
      for (int r = 0; r < 8; r++)
        dst[(size_t)(tbase + rowoff + r) * DH_ + dh] = (_Float16)(acc[r] + bb);
    } else {
      _Float16* dst = vT_out + head_base * DH_ * T_ + (size_t)dh * T_;
#pragma unroll
      for (int r = 0; r < 8; r++)
        dst[tbase + rowoff + r] = (_Float16)(acc[r] + bb);
    }
  }
}

// ---------------- 4) Flash attention: one wave per (head, 32-query tile) -------
// 32 queries/wave: K and V^T tiles are reused by two score / two PV WMMAs.
// Next chunk's K/V tiles are prefetched into registers before the softmax VALU
// work of the current chunk so loads overlap exp/max/sum arithmetic.
__global__ void wa_flash_attn(const _Float16* __restrict__ q_arr,
                              const _Float16* __restrict__ k_arr,
                              const _Float16* __restrict__ vT_arr,
                              _Float16* __restrict__ o_out) {    // [S][B][T][D]
  int wid = blockIdx.x * 8 + (threadIdx.x >> 5);
  int qtile = wid & (T_ / 32 - 1);       // 64 tiles of 32 queries
  int head = wid >> 6;
  if (head >= S_ * B_ * H_) return;
  int lane = threadIdx.x & 31;

  const _Float16* qh = q_arr + (size_t)head * T_ * DH_;
  const _Float16* kh = k_arr + (size_t)head * T_ * DH_;
  const _Float16* vh = vT_arr + (size_t)head * DH_ * T_;
  int qbase = qtile * 32;

  // Q^T B-tiles (K=dh padded to 32): lanes 0..15 = real dh 0..15, lanes 16..31 = pad.
  // Fold 1/sqrt(DH)=0.25 into q.
  v16h bq0, bq1;
#pragma unroll
  for (int i = 0; i < 16; i++) { bq0[i] = (_Float16)0.f; bq1[i] = (_Float16)0.f; }
  if (lane < 16) {
    const _Float16* p0 = qh + (size_t)(qbase + lane) * DH_;
    const _Float16* p1 = qh + (size_t)(qbase + 16 + lane) * DH_;
#pragma unroll
    for (int i = 0; i < 16; i++) {
      bq0[i] = p0[i] * (_Float16)0.25f;
      bq1[i] = p1[i] * (_Float16)0.25f;
    }
  }

  v8f ot0, ot1;
#pragma unroll
  for (int r = 0; r < 8; r++) { ot0[r] = 0.f; ot1[r] = 0.f; }
  float m0 = -1e30f, l0 = 0.f, m1 = -1e30f, l1 = 0.f;

  // prologue loads for chunk 0
  v16h ka0 = loadK_A(kh, 0, lane);
  v16h ka1 = loadK_A(kh, 16, lane);
  v16h va  = loadV_A(vh, 0, lane);

  for (int c0 = 0; c0 < T_; c0 += 32) {
    // prefetch next chunk (wrap-around keeps the loop branch-free / EXEC full)
    int cn = (c0 + 32) & (T_ - 1);
    v16h nka0 = loadK_A(kh, cn, lane);
    v16h nka1 = loadK_A(kh, cn + 16, lane);
    v16h nva  = loadV_A(vh, cn, lane);

    // S^T tiles: lane = query, 4 WMMAs reuse the two K tiles
    v8f s00, s01, s10, s11;
#pragma unroll
    for (int r = 0; r < 8; r++) { s00[r] = 0.f; s01[r] = 0.f; s10[r] = 0.f; s11[r] = 0.f; }
    s00 = wmma_f16(ka0, bq0, s00);
    s01 = wmma_f16(ka1, bq0, s01);
    s10 = wmma_f16(ka0, bq1, s10);
    s11 = wmma_f16(ka1, bq1, s11);

    // online softmax (VALU) overlaps the prefetch loads
    v16h pb0 = softmax_update(s00, s01, m0, l0, ot0, lane);
    v16h pb1 = softmax_update(s10, s11, m1, l1, ot1, lane);

    // O^T += V^T * P^T (V tile reused twice)
    ot0 = wmma_f16(va, pb0, ot0);
    ot1 = wmma_f16(va, pb1, ot1);

    ka0 = nka0; ka1 = nka1; va = nva;
  }

  // epilogue: normalize and write o[s][b][t][h*16+dh]
  int h = head % H_;
  int bb = (head / H_) % B_;
  int s = head / (H_ * B_);
  int dh0 = (lane & 16) ? 8 : 0;
  float inv0 = 1.0f / l0;
  float inv1 = 1.0f / l1;
  {
    int query = qbase + (lane & 15);
    _Float16* dst = o_out + (((size_t)(s * B_ + bb) * T_) + query) * D_ + h * 16 + dh0;
#pragma unroll
    for (int r = 0; r < 8; r++) dst[r] = (_Float16)(ot0[r] * inv0);
  }
  {
    int query = qbase + 16 + (lane & 15);
    _Float16* dst = o_out + (((size_t)(s * B_ + bb) * T_) + query) * D_ + h * 16 + dh0;
#pragma unroll
    for (int r = 0; r < 8; r++) dst[r] = (_Float16)(ot1[r] * inv1);
  }
}

// ---------------- 5) Output projection (WMMA): [S*B*T x 128] x [128 x 128] ------
__global__ void wa_out_proj(const _Float16* __restrict__ o_in,   // [S*B*T][D]
                            const _Float16* __restrict__ owh,    // [D][D] (n,d)
                            const float* __restrict__ out_b,
                            float* __restrict__ o_proj) {        // [S*B*T][D]
  int wid = blockIdx.x * 8 + (threadIdx.x >> 5);
  int ntile = wid & 7;
  int ttile = wid >> 3;
  if (ttile >= S_ * BT_ / 16) return;
  int lane = threadIdx.x & 31;

  const _Float16* A = o_in + (size_t)ttile * 16 * D_;
  const _Float16* Bt = owh + (size_t)ntile * 16 * D_;
  v8f acc;
#pragma unroll
  for (int r = 0; r < 8; r++) acc[r] = 0.f;
#pragma unroll
  for (int kc = 0; kc < D_; kc += 32) {
    v16h a = loadA16x32(A, D_, kc);
    v16h b = loadB32x16(Bt, D_, kc);
    acc = wmma_f16(a, b, acc);
  }
  int n = ntile * 16 + (lane & 15);
  float bb = out_b[n];
  float* dst = o_proj + ((size_t)ttile * 16 + ((lane & 16) ? 8 : 0)) * D_ + n;
#pragma unroll
  for (int r = 0; r < 8; r++) dst[(size_t)r * D_] = acc[r] + bb;
}

// ---------------- 6) gating ----------------
__global__ void wa_global_feat(const float* __restrict__ o_proj, float* __restrict__ feat) {
  int b = blockIdx.x >> 7;                 // B*D blocks
  int d = blockIdx.x & (D_ - 1);
  float acc = 0.f;
  for (int i = threadIdx.x; i < S_ * T_; i += blockDim.x) {
    int s = i / T_;
    int t = i - s * T_;
    acc += o_proj[(((size_t)s * B_ + b) * T_ + t) * D_ + d];
  }
  __shared__ float red[256];
  red[threadIdx.x] = acc;
  __syncthreads();
  for (int off = 128; off > 0; off >>= 1) {
    if (threadIdx.x < off) red[threadIdx.x] += red[threadIdx.x + off];
    __syncthreads();
  }
  if (threadIdx.x == 0) feat[b * D_ + d] = red[0] / (float)(S_ * T_);
}

__global__ void wa_gate(const float* __restrict__ feat, const float* __restrict__ gw,
                        const float* __restrict__ gb, float* __restrict__ gate) {
  __shared__ float red[128];
  __shared__ float logit[B_ * S_];
  int tid = threadIdx.x;                    // 128 threads
  for (int b = 0; b < B_; b++) {
    for (int s = 0; s < S_; s++) {
      red[tid] = feat[b * D_ + tid] * gw[s * D_ + tid];
      __syncthreads();
      for (int off = 64; off > 0; off >>= 1) {
        if (tid < off) red[tid] += red[tid + off];
        __syncthreads();
      }
      if (tid == 0) logit[b * S_ + s] = red[0] + gb[s];
      __syncthreads();
    }
  }
  if (tid == 0) {
    for (int b = 0; b < B_; b++) {
      float mx = -1e30f;
      for (int s = 0; s < S_; s++) mx = fmaxf(mx, logit[b * S_ + s]);
      float sum = 0.f, e[S_];
      for (int s = 0; s < S_; s++) { e[s] = __expf(logit[b * S_ + s] - mx); sum += e[s]; }
      for (int s = 0; s < S_; s++) gate[b * S_ + s] = e[s] / sum;
    }
  }
}

__global__ void wa_combine(const float* __restrict__ o_proj, const float* __restrict__ gate,
                           float* __restrict__ Z) {
  int idx = blockIdx.x * blockDim.x + threadIdx.x;   // B*T*D
  if (idx >= B_ * T_ * D_) return;
  int d = idx % D_;
  int t = (idx / D_) % T_;
  int b = idx / (D_ * T_);
  float acc = 0.f;
#pragma unroll
  for (int s = 0; s < S_; s++)
    acc += o_proj[(((size_t)s * B_ + b) * T_ + t) * D_ + d] * gate[b * S_ + s];
  Z[idx] = acc;
}

// ---------------- host launcher ----------------
extern "C" void kernel_launch(void* const* d_in, const int* in_sizes, int n_in,
                              void* d_out, int out_size, void* d_ws, size_t ws_size,
                              hipStream_t stream) {
  const float* x     = (const float*)d_in[0];
  const float* Wq    = (const float*)d_in[1];
  const float* bq    = (const float*)d_in[2];
  const float* Wk    = (const float*)d_in[3];
  const float* bk    = (const float*)d_in[4];
  const float* Wv    = (const float*)d_in[5];
  const float* bv    = (const float*)d_in[6];
  const float* in_w  = (const float*)d_in[7];
  const float* in_b  = (const float*)d_in[8];
  const float* out_w = (const float*)d_in[9];
  const float* out_b = (const float*)d_in[10];
  const float* gw    = (const float*)d_in[11];
  const float* gb    = (const float*)d_in[12];
  float* Z = (float*)d_out;

  char* base = (char*)d_ws;
  size_t off = 0;
  auto carve = [&](size_t bytes) -> char* {
    char* p = base + off;
    off += (bytes + 255) & ~(size_t)255;
    return p;
  };
  _Float16* Wdet   = (_Float16*)carve((size_t)S_ * BT_ * C_ * 2);       // 1.5 MB
  float*    a0     = (float*)   carve((size_t)BT_ * C_ * 4);            // 1 MB
  float*    a1     = (float*)   carve((size_t)BT_ * C_ * 4);            // 1 MB
  _Float16* Amat   = (_Float16*)carve((size_t)3 * S_ * D_ * C_ * 2);
  float*    bias   = (float*)   carve((size_t)3 * S_ * D_ * 4);
  _Float16* owh    = (_Float16*)carve((size_t)D_ * D_ * 2);
  _Float16* q_arr  = (_Float16*)carve((size_t)S_ * B_ * H_ * T_ * DH_ * 2);  // 3 MB
  _Float16* k_arr  = (_Float16*)carve((size_t)S_ * B_ * H_ * T_ * DH_ * 2);  // 3 MB
  _Float16* vT_arr = (_Float16*)carve((size_t)S_ * B_ * H_ * DH_ * T_ * 2);  // 3 MB
  _Float16* o_att  = (_Float16*)carve((size_t)S_ * BT_ * D_ * 2);            // 3 MB
  float*    o_proj = (float*)   carve((size_t)S_ * BT_ * D_ * 4);            // 6.3 MB
  float*    feat   = (float*)   carve((size_t)B_ * D_ * 4);
  float*    gate   = (float*)   carve((size_t)B_ * S_ * 4);
  (void)ws_size; (void)in_sizes; (void)n_in; (void)out_size;

  // 1) SWT: 3 levels; detail level j goes to scale slot s = S-1-j (pywt swt order)
  int nswt = B_ * T_ * C_;
  dim3 bswt(256), gswt((nswt + 255) / 256);
  wa_swt_level<<<gswt, bswt, 0, stream>>>(x,  a0, Wdet + (size_t)2 * BT_ * C_, 1);
  wa_swt_level<<<gswt, bswt, 0, stream>>>(a0, a1, Wdet + (size_t)1 * BT_ * C_, 2);
  wa_swt_level<<<gswt, bswt, 0, stream>>>(a1, a0, Wdet + (size_t)0 * BT_ * C_, 4);

  // 2) fold weights
  wa_fold_weights<<<dim3(3 * S_ * D_), dim3(C_), 0, stream>>>(Wq, bq, Wk, bk, Wv, bv,
                                                              in_w, in_b, Amat, bias);
  wa_cvt_outw<<<dim3((D_ * D_ + 255) / 256), dim3(256), 0, stream>>>(out_w, owh);

  // 3) QKV GEMM: S * (BT/16) * 8 = 6144 waves -> 768 blocks of 256 threads
  wa_qkv_gemm<<<dim3(768), dim3(256), 0, stream>>>(Wdet, Amat, bias, q_arr, k_arr, vT_arr);

  // 4) flash attention: S*B*H*(T/32) = 3072 waves -> 384 blocks of 256 threads
  wa_flash_attn<<<dim3(384), dim3(256), 0, stream>>>(q_arr, k_arr, vT_arr, o_att);

  // 5) out projection: (S*BT/16)*8 = 6144 waves -> 768 blocks
  wa_out_proj<<<dim3(768), dim3(256), 0, stream>>>(o_att, owh, out_b, o_proj);

  // 6) gating + combine
  wa_global_feat<<<dim3(B_ * D_), dim3(256), 0, stream>>>(o_proj, feat);
  wa_gate<<<dim3(1), dim3(128), 0, stream>>>(feat, gw, gb, gate);
  wa_combine<<<dim3((B_ * T_ * D_ + 255) / 256), dim3(256), 0, stream>>>(o_proj, gate, Z);
}
